// MultiHeadAttention_24352464570159
// MI455X (gfx1250) — matrix operand order repaired and checked by reading, not compiled, not run on hardware
//
#include <hip/hip_runtime.h>

// ---------------------------------------------------------------------------
// Types for CDNA5 WMMA (wave32, 16x16x32 bf16 -> f32)
// ---------------------------------------------------------------------------
typedef __bf16 bf16_t;
typedef __attribute__((ext_vector_type(16))) __bf16 v16bf;
typedef __attribute__((ext_vector_type(8)))  __bf16 v8bf;
typedef __attribute__((ext_vector_type(8)))  float  v8f;

// GCC-style 16-byte int vector, matching the async-LDS builtin's pointee type.
typedef int v4i_t __attribute__((vector_size(16)));
typedef __attribute__((address_space(1))) v4i_t gv4i_t;   // global
typedef __attribute__((address_space(3))) v4i_t lv4i_t;   // LDS

// ---------------------------------------------------------------------------
// CDNA5 async global->LDS copy (ASYNCcnt path), with safe fallback.
// ---------------------------------------------------------------------------
#if defined(__gfx1250__) && __has_builtin(__builtin_amdgcn_global_load_async_to_lds_b128)
#define HAVE_ASYNC_LDS 1
#else
#define HAVE_ASYNC_LDS 0
#endif

__device__ __forceinline__ void copy16_async(bf16_t* lds_dst, const bf16_t* gsrc) {
#if HAVE_ASYNC_LDS
    __builtin_amdgcn_global_load_async_to_lds_b128(
        (gv4i_t*)gsrc, (lv4i_t*)lds_dst, /*offset=*/0, /*cpol=*/0);
#else
    *(v8bf*)lds_dst = *(const v8bf*)gsrc;
#endif
}

__device__ __forceinline__ void wait_async_all() {
#if HAVE_ASYNC_LDS
#if __has_builtin(__builtin_amdgcn_s_wait_asynccnt)
    __builtin_amdgcn_s_wait_asynccnt(0);
#else
    asm volatile("s_wait_asynccnt 0" ::: "memory");
#endif
#endif
}

__device__ __forceinline__ v8f wmma_bf16(v16bf a, v16bf b, v8f c) {
    return __builtin_amdgcn_wmma_f32_16x16x32_bf16(
        /*neg_a=*/false, a, /*neg_b=*/false, b,
        /*c_mod=*/(short)0, c, /*reuse_a=*/false, /*reuse_b=*/false);
}

// A fragment (16x32, MxK) from LDS tile stored row-major [rows][ld] (K contig).
// ISA 7.12.2: lane(hi) holds K {hi*8+0..7} U {16+hi*8+0..7} of row (lane&15).
__device__ __forceinline__ v16bf load_frag_a(const bf16_t* lds, int row0, int k0,
                                             int ld, int lane) {
    const int r = lane & 15, hi = lane >> 4;
    const bf16_t* p = lds + (row0 + r) * ld + k0 + hi * 8;
    union { v16bf f; v8bf h[2]; } u;
    u.h[0] = *(const v8bf*)p;
    u.h[1] = *(const v8bf*)(p + 16);
    return u.f;
}

// B fragment (32x16, KxN) from LDS tile stored row-major [N][ld] (K contig):
// element B[k][n] at lds[n*ld + k]; lane n=lane&15 holds K {hi*8+0..7} U {16+hi*8+0..7}.
__device__ __forceinline__ v16bf load_frag_b(const bf16_t* lds, int n0, int k0,
                                             int ld, int lane) {
    const int n = lane & 15, hi = lane >> 4;
    const bf16_t* p = lds + (n0 + n) * ld + k0 + hi * 8;
    union { v16bf f; v8bf h[2]; } u;
    u.h[0] = *(const v8bf*)p;
    u.h[1] = *(const v8bf*)(p + 16);
    return u.f;
}

// ---------------------------------------------------------------------------
// Pack f32 -> bf16 (element count multiple of 2048).
// ---------------------------------------------------------------------------
__global__ __launch_bounds__(256) void mha_pack_bf16(const float* __restrict__ src,
                                                     bf16_t* __restrict__ dst) {
    const size_t i = ((size_t)blockIdx.x * 256 + threadIdx.x) * 8;
    float4 a = *(const float4*)(src + i);
    float4 b = *(const float4*)(src + i + 4);
    union { v8bf v; bf16_t e[8]; } u;
    u.e[0] = (bf16_t)a.x; u.e[1] = (bf16_t)a.y; u.e[2] = (bf16_t)a.z; u.e[3] = (bf16_t)a.w;
    u.e[4] = (bf16_t)b.x; u.e[5] = (bf16_t)b.y; u.e[6] = (bf16_t)b.z; u.e[7] = (bf16_t)b.w;
    *(v8bf*)(dst + i) = u.v;
}

// ---------------------------------------------------------------------------
// GEMM: C[M,N] = A[M,K] * W[N,K]^T, bf16 in / f32 out, f32 WMMA accumulate.
// Block tile 128x128, 256 threads = 8 waves (2x4), wave tile 64x32.
// K-step 64, double-buffered LDS, async global->LDS copies.
// N is a template constant so epilogue store offsets are immediates.
// ---------------------------------------------------------------------------
template <int N>
__global__ __launch_bounds__(256) void mha_gemm_bf16(const bf16_t* __restrict__ A,
                                                     const bf16_t* __restrict__ W,
                                                     float* __restrict__ C,
                                                     int M, int K) {
    __shared__ __align__(16) bf16_t As[2][128 * 64];   // 16KB x2
    __shared__ __align__(16) bf16_t Ws[2][128 * 64];   // 16KB x2
    const int tid = threadIdx.x, lane = tid & 31, wave = tid >> 5;
    const int wm = (wave >> 2) * 64;   // 0 or 64
    const int wn = (wave & 3) * 32;    // 0,32,64,96
    const int mBase = blockIdx.y * 128, nBase = blockIdx.x * 128;

    auto copy_tile = [&](int buf, int kt) {
#pragma unroll
        for (int j = 0; j < 4; ++j) {
            int c = tid + 256 * j;            // 1024 x 16B chunks per operand
            int row = c >> 3, col = (c & 7) * 8;
            copy16_async(&As[buf][row * 64 + col],
                         A + (size_t)(mBase + row) * K + kt + col);
            copy16_async(&Ws[buf][row * 64 + col],
                         W + (size_t)(nBase + row) * K + kt + col);
        }
    };

    v8f acc[4][2] = {};
    copy_tile(0, 0);
    wait_async_all();
    __syncthreads();

    int buf = 0;
    for (int kt = 0; kt < K; kt += 64) {
        if (kt + 64 < K) copy_tile(buf ^ 1, kt + 64);   // overlap with WMMA below
#pragma unroll
        for (int ks = 0; ks < 64; ks += 32) {
            v16bf b0 = load_frag_b(Ws[buf], wn,      ks, 64, lane);
            v16bf b1 = load_frag_b(Ws[buf], wn + 16, ks, 64, lane);
#pragma unroll
            for (int mi = 0; mi < 4; ++mi) {
                v16bf a = load_frag_a(As[buf], wm + mi * 16, ks, 64, lane);
                acc[mi][0] = wmma_bf16(a, b0, acc[mi][0]);
                acc[mi][1] = wmma_bf16(a, b1, acc[mi][1]);
            }
        }
        wait_async_all();
        __syncthreads();
        buf ^= 1;
    }

    const int hi = lane >> 4, ln = lane & 15;
    float* base = C + (size_t)(mBase + wm + hi * 8) * N + nBase + wn + ln;
#pragma unroll
    for (int mi = 0; mi < 4; ++mi)
#pragma unroll
        for (int ni = 0; ni < 2; ++ni)
#pragma unroll
            for (int e = 0; e < 8; ++e)
                base[(size_t)((mi * 16 + e) * N + ni * 16)] = acc[mi][ni][e];
}

// ---------------------------------------------------------------------------
// RoPE for Q: read Qf [B*S, 2048] (row=b*2048+s, col=h*64+d), write bf16
// Q' in [B, 32, S, 64]. inv_freq[j] = 10000^(-j/32), j = d & 31.
// ---------------------------------------------------------------------------
__global__ __launch_bounds__(256) void mha_rope_q(const float* __restrict__ Qf,
                                                  bf16_t* __restrict__ Qbf) {
    const size_t idx = (size_t)blockIdx.x * 256 + threadIdx.x;  // < 8192*2048
    const int row = (int)(idx >> 11);
    const int col = (int)(idx & 2047);
    const int h = col >> 6, d = col & 63;
    const int b = row >> 11, s = row & 2047;
    const int j = d & 31;
    float ang = (float)s * __powf(10000.0f, -(float)j * (1.0f / 32.0f));
    float c = __cosf(ang), sn = __sinf(ang);
    float q = Qf[idx];
    float rot = (d < 32) ? -Qf[idx + 32] : Qf[idx - 32];
    Qbf[(((size_t)b * 32 + h) * 2048 + s) * 64 + d] = (bf16_t)(q * c + rot * sn);
}

// ---------------------------------------------------------------------------
// RoPE for K (+ V passthrough): read Kf/Vf [B*S, 512], write bf16 K'/V in
// [B, 8, S, 64] for attention, plus f32 K'/V to d_out (kv_cache tuple).
// ---------------------------------------------------------------------------
__global__ __launch_bounds__(256) void mha_rope_kv(const float* __restrict__ Kf,
                                                   const float* __restrict__ Vf,
                                                   bf16_t* __restrict__ Kbf,
                                                   bf16_t* __restrict__ Vbf,
                                                   float* __restrict__ outK,
                                                   float* __restrict__ outV) {
    const size_t idx = (size_t)blockIdx.x * 256 + threadIdx.x;  // < 8192*512
    const int row = (int)(idx >> 9);
    const int col = (int)(idx & 511);
    const int kvh = col >> 6, d = col & 63;
    const int b = row >> 11, s = row & 2047;
    const int j = d & 31;
    float ang = (float)s * __powf(10000.0f, -(float)j * (1.0f / 32.0f));
    float c = __cosf(ang), sn = __sinf(ang);
    float k = Kf[idx];
    float rot = (d < 32) ? -Kf[idx + 32] : Kf[idx - 32];
    float kout = k * c + rot * sn;
    float v = Vf[idx];
    const size_t oidx = (((size_t)b * 8 + kvh) * 2048 + s) * 64 + d;
    Kbf[oidx] = (bf16_t)kout;
    Vbf[oidx] = (bf16_t)v;
    outK[oidx] = kout;
    outV[oidx] = v;
}

// ---------------------------------------------------------------------------
// Causal flash attention, GQA. One block = one KV group (4 query heads) x one
// 64-query tile. 512 threads = 16 waves; wave = (head-in-group, 16 q rows).
// K/V tiles loaded once per key block and reused by all 4 heads.
// Q LDS is dead once fragments are in registers -> P scratch overlays it.
// ---------------------------------------------------------------------------
__global__ __launch_bounds__(512) void mha_flash_attn(const bf16_t* __restrict__ Qbf,
                                                      const bf16_t* __restrict__ Kbf,
                                                      const bf16_t* __restrict__ Vbf,
                                                      const float* __restrict__ temp,
                                                      bf16_t* __restrict__ attnOut) {
    __shared__ __align__(16) bf16_t Qps[4 * 64 * 64];  // 32KB: Q tiles, then P scratch
    __shared__ __align__(16) bf16_t Ks[64 * 64];       // 8KB  [key][dh]
    __shared__ __align__(16) bf16_t Vs[64 * 64];       // 8KB  [dh][key] (transposed)

    const int tid = threadIdx.x, lane = tid & 31, wave = tid >> 5;
    const int b = blockIdx.y >> 3, kvh = blockIdx.y & 7;
    const int hg = wave >> 2;              // head within kv group: 0..3
    const int h = kvh * 4 + hg;
    const int qrow0 = (wave & 3) * 16;     // q rows within 64-q tile
    const int qbase = blockIdx.x * 64;
    const float scale = 0.125f / fmaxf(fabsf(temp[0]), 1e-6f);  // rsqrt(64)/max(|T|,eps)

    // Load the 4 heads' Q tiles (each a contiguous 4096-bf16 block).
#pragma unroll
    for (int j = 0; j < 4; ++j) {
        int c = tid + 512 * j;             // 0..2047 chunks of 8 bf16
        int hh = c >> 9, cc = c & 511;
        const bf16_t* Qg = Qbf + (((size_t)b * 32 + kvh * 4 + hh) * 2048 + qbase) * 64;
        copy16_async(&Qps[hh * 4096 + cc * 8], Qg + cc * 8);
    }
    wait_async_all();
    __syncthreads();

    const v16bf aq0 = load_frag_a(Qps + hg * 4096, qrow0, 0,  64, lane);
    const v16bf aq1 = load_frag_a(Qps + hg * 4096, qrow0, 32, 64, lane);
    bf16_t* Pw = Qps + wave * (16 * 64);   // per-wave P tile (overlays dead Q)

    float m[8], l[8];
    v8f o[4] = {};
#pragma unroll
    for (int e = 0; e < 8; ++e) { m[e] = -1e30f; l[e] = 0.0f; }
    const int hi = lane >> 4, ln = lane & 15;

    const bf16_t* Kbase = Kbf + ((size_t)b * 8 + kvh) * 2048 * 64;
    const bf16_t* Vbase = Vbf + ((size_t)b * 8 + kvh) * 2048 * 64;

    for (int kb = 0; kb <= (int)blockIdx.x; ++kb) {
        __syncthreads();  // previous iteration's LDS reads done before overwrite
        // K tile: 512 x 16B chunks, one per thread, async path.
        copy16_async(&Ks[tid * 8], Kbase + (size_t)kb * 4096 + tid * 8);
        // V tile transposed into [dh][key]: 1 x b128 load + 8 ds_store_b16.
        {
            const int key = tid >> 3, d0 = (tid & 7) * 8;
            v8bf v = *(const v8bf*)(Vbase + (size_t)kb * 4096 + key * 64 + d0);
#pragma unroll
            for (int jj = 0; jj < 8; ++jj) Vs[(d0 + jj) * 64 + key] = v[jj];
        }
        wait_async_all();
        __syncthreads();

        // scores = Q * K^T (16 queries x 64 keys per wave)
        v8f sc[4];
#pragma unroll
        for (int n = 0; n < 4; ++n) {
            sc[n] = v8f{};
            sc[n] = wmma_bf16(aq0, load_frag_b(Ks, n * 16, 0,  64, lane), sc[n]);
            sc[n] = wmma_bf16(aq1, load_frag_b(Ks, n * 16, 32, 64, lane), sc[n]);
        }

        const bool diag = (kb == (int)blockIdx.x);
        float mx[8];
#pragma unroll
        for (int e = 0; e < 8; ++e) mx[e] = -1e30f;
#pragma unroll
        for (int n = 0; n < 4; ++n)
#pragma unroll
            for (int e = 0; e < 8; ++e) {
                float s = sc[n][e] * scale;
                if (diag && (n * 16 + ln > qrow0 + e + hi * 8)) s = -1e30f;
                sc[n][e] = s;
                mx[e] = fmaxf(mx[e], s);
            }
        // Row reduce across the 16 lanes of this half-wave (rows live per half).
#pragma unroll
        for (int off = 1; off < 16; off <<= 1)
#pragma unroll
            for (int e = 0; e < 8; ++e)
                mx[e] = fmaxf(mx[e], __shfl_xor(mx[e], off, 32));

        float f[8], rs[8];
#pragma unroll
        for (int e = 0; e < 8; ++e) {
            float mn = fmaxf(m[e], mx[e]);
            f[e] = __expf(m[e] - mn);
            m[e] = mn;
            rs[e] = 0.0f;
        }
#pragma unroll
        for (int n = 0; n < 4; ++n)
#pragma unroll
            for (int e = 0; e < 8; ++e) {
                float p = __expf(sc[n][e] - m[e]);
                rs[e] += p;
                Pw[(e + hi * 8) * 64 + n * 16 + ln] = (bf16_t)p;
            }
#pragma unroll
        for (int off = 1; off < 16; off <<= 1)
#pragma unroll
            for (int e = 0; e < 8; ++e)
                rs[e] += __shfl_xor(rs[e], off, 32);
#pragma unroll
        for (int e = 0; e < 8; ++e) l[e] = l[e] * f[e] + rs[e];
#pragma unroll
        for (int n = 0; n < 4; ++n)
#pragma unroll
            for (int e = 0; e < 8; ++e) o[n][e] *= f[e];

        // Wave-local LDS RAW: this wave's ds_stores must land before P reads.
        asm volatile("s_wait_dscnt 0" ::: "memory");
        v16bf ap0 = load_frag_a(Pw, 0, 0,  64, lane);
        v16bf ap1 = load_frag_a(Pw, 0, 32, 64, lane);
#pragma unroll
        for (int n = 0; n < 4; ++n) {
            o[n] = wmma_bf16(ap0, load_frag_b(Vs, n * 16, 0,  64, lane), o[n]);
            o[n] = wmma_bf16(ap1, load_frag_b(Vs, n * 16, 32, 64, lane), o[n]);
        }
    }

    // out /= l ; emit bf16 in [B, S, 32, 64] so rows match the (B*S, 2048) GEMM A.
#pragma unroll
    for (int n = 0; n < 4; ++n)
#pragma unroll
        for (int e = 0; e < 8; ++e) {
            int q = qbase + qrow0 + e + hi * 8;
            int dh = n * 16 + ln;
            attnOut[(((size_t)b * 2048 + q) * 32 + h) * 64 + dh] =
                (bf16_t)(o[n][e] / l[e]);
        }
}

// ---------------------------------------------------------------------------
// Host launcher.
// ---------------------------------------------------------------------------
extern "C" void kernel_launch(void* const* d_in, const int* in_sizes, int n_in,
                              void* d_out, int out_size, void* d_ws, size_t ws_size,
                              hipStream_t stream) {
    (void)in_sizes; (void)n_in; (void)out_size; (void)ws_size;
    const float* x    = (const float*)d_in[0];
    const float* Wq   = (const float*)d_in[1];
    const float* Wk   = (const float*)d_in[2];
    const float* Wv   = (const float*)d_in[3];
    const float* Wo   = (const float*)d_in[4];
    const float* temp = (const float*)d_in[5];

    float* y    = (float*)d_out;
    float* outK = y + (size_t)16777216;
    float* outV = outK + (size_t)4194304;

    char* ws = (char*)d_ws;
    bf16_t* xbf  = (bf16_t*)(ws);                      //  33,554,432 B
    bf16_t* Wqbf = (bf16_t*)(ws + 33554432ull);        //   8,388,608 B
    bf16_t* Wkbf = (bf16_t*)(ws + 41943040ull);        //   2,097,152 B
    bf16_t* Wvbf = (bf16_t*)(ws + 44040192ull);        //   2,097,152 B
    bf16_t* Wobf = (bf16_t*)(ws + 46137344ull);        //   8,388,608 B
    float*  Qf   = (float*) (ws + 54525952ull);        //  67,108,864 B
    float*  Kf   = (float*) (ws + 121634816ull);       //  16,777,216 B
    float*  Vf   = (float*) (ws + 138412032ull);       //  16,777,216 B
    bf16_t* Qbf  = (bf16_t*)(ws + 155189248ull);       //  33,554,432 B
    bf16_t* Kbf  = (bf16_t*)(ws + 188743680ull);       //   8,388,608 B
    bf16_t* Vbf  = (bf16_t*)(ws + 197132288ull);       //   8,388,608 B  (end ~206MB)
    bf16_t* attnOut = xbf;  // x is dead after the projections -> reuse

    // f32 -> bf16 packs.
    mha_pack_bf16<<<16777216 / 2048, 256, 0, stream>>>(x,  xbf);
    mha_pack_bf16<<<4194304  / 2048, 256, 0, stream>>>(Wq, Wqbf);
    mha_pack_bf16<<<1048576  / 2048, 256, 0, stream>>>(Wk, Wkbf);
    mha_pack_bf16<<<1048576  / 2048, 256, 0, stream>>>(Wv, Wvbf);
    mha_pack_bf16<<<4194304  / 2048, 256, 0, stream>>>(Wo, Wobf);

    // Projections: M = B*S = 8192, K = 2048.
    mha_gemm_bf16<2048><<<dim3(16, 64), 256, 0, stream>>>(xbf, Wqbf, Qf, 8192, 2048);
    mha_gemm_bf16<512> <<<dim3(4, 64),  256, 0, stream>>>(xbf, Wkbf, Kf, 8192, 2048);
    mha_gemm_bf16<512> <<<dim3(4, 64),  256, 0, stream>>>(xbf, Wvbf, Vf, 8192, 2048);

    // RoPE + layout change + kv_cache outputs.
    mha_rope_q <<<(8192 * 2048) / 256, 256, 0, stream>>>(Qf, Qbf);
    mha_rope_kv<<<(8192 * 512)  / 256, 256, 0, stream>>>(Kf, Vf, Kbf, Vbf, outK, outV);

    // Flash attention: grid (S/64, B*KV).
    mha_flash_attn<<<dim3(32, 32), 512, 0, stream>>>(Qbf, Kbf, Vbf, temp, attnOut);

    // Output projection into y.
    mha_gemm_bf16<2048><<<dim3(16, 64), 256, 0, stream>>>(attnOut, Wobf, y, 8192, 2048);
}
